// DA_HGNN_35948876268092
// MI455X (gfx1250) — 16-bit and 8-bit matrix operand order repaired and checked by reading, compiled, not run on hardware
//
#include <hip/hip_runtime.h>
#include <hip/hip_bf16.h>

#define NN 4096
#define FF 512
#define HIDW 256
#define DD 8
#define NHEADS 4
#define SIG 0.5f

typedef __attribute__((ext_vector_type(16))) __bf16 v16bf;
typedef __attribute__((ext_vector_type(8))) float v8f;
typedef __attribute__((ext_vector_type(8))) int v8i;
typedef __attribute__((ext_vector_type(8))) unsigned int v8u;

typedef int i32x4 __attribute__((vector_size(16)));
using g_i32x4 = __attribute__((address_space(1))) i32x4;
using l_i32x4 = __attribute__((address_space(3))) i32x4;

#if defined(__has_builtin)
#if __has_builtin(__builtin_amdgcn_global_load_async_to_lds_b128)
#define HAVE_ASYNC_LDS 1
#endif
#endif
#ifndef HAVE_ASYNC_LDS
#define HAVE_ASYNC_LDS 0
#endif

__device__ __forceinline__ unsigned short f2bf(float f) {
  unsigned int u = __float_as_uint(f);
  u += 0x7FFFu + ((u >> 16) & 1u);
  return (unsigned short)(u >> 16);
}
__device__ __forceinline__ float lrelu(float x) { return x > 0.f ? x : 0.2f * x; }
__device__ __forceinline__ float eluf(float x)  { return x > 0.f ? x : (expf(x) - 1.f); }

__device__ __forceinline__ void cp16(void* l, const void* g) {
#if HAVE_ASYNC_LDS
  __builtin_amdgcn_global_load_async_to_lds_b128((g_i32x4*)g, (l_i32x4*)l, 0, 0);
#else
  *reinterpret_cast<uint4*>(l) = *reinterpret_cast<const uint4*>(g);
#endif
}
__device__ __forceinline__ void st16z(void* l) {
  uint4 z = {0u, 0u, 0u, 0u};
  *reinterpret_cast<uint4*>(l) = z;
}
__device__ __forceinline__ void wait_stage() {
#if HAVE_ASYNC_LDS
#if __has_builtin(__builtin_amdgcn_s_wait_asynccnt)
  __builtin_amdgcn_s_wait_asynccnt(0);
#endif
#endif
  __syncthreads();
}

template <bool TRANSB>
__global__ __launch_bounds__(256) void gemm_bf16_kernel(
    const unsigned short* __restrict__ A, const unsigned short* __restrict__ B,
    float* __restrict__ C, int M, int N, int K) {
  __shared__ unsigned short sA[2][64][64];
  __shared__ unsigned short sB[2][8192];
  const int tid = threadIdx.x;
  const int lane = tid & 31, wid = tid >> 5;
  const int half = lane >> 4, mn = lane & 15;
  const int mW = (wid & 3) << 4;
  const int nW = (wid >> 2) << 6;
  const int m0 = blockIdx.y * 64, n0 = blockIdx.x * 128;
  v8f z = {0.f, 0.f, 0.f, 0.f, 0.f, 0.f, 0.f, 0.f};
  v8f acc[4] = {z, z, z, z};

  auto stage = [&](int kk, int buf) {
    for (int c = tid; c < 512; c += 256) {
      const int row = c >> 3, kc = (c & 7) << 3;
      const int gr = m0 + row, gk = kk + kc;
      void* l = &sA[buf][row][kc];
      if (gr < M && (gk + 7) < K) cp16(l, A + (size_t)gr * K + gk);
      else st16z(l);
    }
    if (TRANSB) {
      for (int c = tid; c < 1024; c += 256) {
        const int row = c >> 3, kc = (c & 7) << 3;
        const int gr = n0 + row, gk = kk + kc;
        void* l = &sB[buf][row * 64 + kc];
        if (gr < N && (gk + 7) < K) cp16(l, B + (size_t)gr * K + gk);
        else st16z(l);
      }
    } else {
      for (int c = tid; c < 1024; c += 256) {
        const int kr = c >> 4, nb = (c & 15) << 3;
        const int gk = kk + kr, gn = n0 + nb;
        void* l = &sB[buf][kr * 128 + nb];
        if (gk < K && (gn + 7) < N) cp16(l, B + (size_t)gk * N + gn);
        else st16z(l);
      }
    }
  };

  auto compute = [&](int buf) {
    #pragma unroll
    for (int ks = 0; ks < 64; ks += 32) {
      v8u au;
      #pragma unroll
      for (int r = 0; r < 8; ++r) {
        const int ka = ks + 2 * r + ((r & 4) ? 8 : 0) + 8 * half;
        au[r] = *reinterpret_cast<const unsigned int*>(&sA[buf][mW + mn][ka]);
      }
      const v16bf av = __builtin_bit_cast(v16bf, au);
      #pragma unroll
      for (int t = 0; t < 4; ++t) {
        const int col = nW + t * 16 + mn;
        v8u bu;
        #pragma unroll
        for (int r = 0; r < 8; ++r) {
          const int kb = ks + 2 * r + ((r & 4) ? 8 : 0) + 8 * half;
          if (TRANSB) {
            bu[r] = *reinterpret_cast<const unsigned int*>(&sB[buf][col * 64 + kb]);
          } else {
            const unsigned lo = sB[buf][kb * 128 + col];
            const unsigned hi = sB[buf][(kb + 1) * 128 + col];
            bu[r] = lo | (hi << 16);
          }
        }
        acc[t] = __builtin_amdgcn_wmma_f32_16x16x32_bf16(
            false, av, false, __builtin_bit_cast(v16bf, bu), (short)0, acc[t], false, false);
      }
    }
  };

  stage(0, 0);
  int buf = 0;
  for (int kk = 0; kk < K; kk += 64) {
    wait_stage();
    if (kk + 64 < K) stage(kk + 64, buf ^ 1);
    compute(buf);
    __syncthreads();
    buf ^= 1;
  }
  #pragma unroll
  for (int t = 0; t < 4; ++t) {
    #pragma unroll
    for (int p = 0; p < 8; ++p) {
      const int row = m0 + mW + p + 8 * half;
      const int col = n0 + nW + t * 16 + mn;
      if (row < M && col < N) C[(size_t)row * N + col] = acc[t][p];
    }
  }
}

__global__ __launch_bounds__(256) void gemm_iu8_bool_kernel(
    const unsigned char* __restrict__ A, const unsigned char* __restrict__ B,
    unsigned char* __restrict__ D8, unsigned short* __restrict__ Dbf,
    int n, int offdiag) {
  __shared__ unsigned char sA[2][64][64];
  __shared__ unsigned char sB[2][128][64];
  const int tid = threadIdx.x, lane = tid & 31, wid = tid >> 5;
  const int half = lane >> 4, mn = lane & 15;
  const int mW = (wid & 3) << 4;
  const int nW = (wid >> 2) << 6;
  const int m0 = blockIdx.y * 64, n0 = blockIdx.x * 128;
  v8i zi = {0, 0, 0, 0, 0, 0, 0, 0};
  v8i acc[4] = {zi, zi, zi, zi};

  auto stage = [&](int kk, int buf) {
    {
      const int c = tid;
      const int row = c >> 2, kc = (c & 3) << 4;
      cp16(&sA[buf][row][kc], A + (size_t)(m0 + row) * n + kk + kc);
    }
    for (int c = tid; c < 512; c += 256) {
      const int row = c >> 2, kc = (c & 3) << 4;
      cp16(&sB[buf][row][kc], B + (size_t)(n0 + row) * n + kk + kc);
    }
  };

  auto compute = [&](int buf) {
    v8u au;
    #pragma unroll
    for (int r = 0; r < 8; ++r) {
      const int ka = ((r >> 1) << 4) + ((r & 1) << 2) + 8 * half;
      au[r] = *reinterpret_cast<const unsigned int*>(&sA[buf][mW + mn][ka]);
    }
    const v8i av = __builtin_bit_cast(v8i, au);
    #pragma unroll
    for (int t = 0; t < 4; ++t) {
      v8u bu;
      #pragma unroll
      for (int r = 0; r < 8; ++r) {
        const int kb = ((r >> 1) << 4) + ((r & 1) << 2) + 8 * half;
        bu[r] = *reinterpret_cast<const unsigned int*>(&sB[buf][nW + t * 16 + mn][kb]);
      }
      acc[t] = __builtin_amdgcn_wmma_i32_16x16x64_iu8(
          false, av, false, __builtin_bit_cast(v8i, bu), acc[t], false, false);
    }
  };

  stage(0, 0);
  int buf = 0;
  for (int kk = 0; kk < n; kk += 64) {
    wait_stage();
    if (kk + 64 < n) stage(kk + 64, buf ^ 1);
    compute(buf);
    __syncthreads();
    buf ^= 1;
  }
  #pragma unroll
  for (int t = 0; t < 4; ++t) {
    #pragma unroll
    for (int p = 0; p < 8; ++p) {
      const int row = m0 + mW + p + 8 * half;
      const int col = n0 + nW + t * 16 + mn;
      unsigned char v = (acc[t][p] > 0) ? (unsigned char)1 : (unsigned char)0;
      if (offdiag && row == col) v = 0;
      D8[(size_t)row * n + col] = v;
      if (Dbf) Dbf[(size_t)row * n + col] = v ? (unsigned short)0x3F80u : (unsigned short)0;
    }
  }
}

__global__ void k_f32_to_bf16(const float* __restrict__ s, unsigned short* __restrict__ d, int count) {
  int i = blockIdx.x * 256 + threadIdx.x;
  if (i < count) d[i] = f2bf(s[i]);
}

__global__ void k_row_sumsq(const float* __restrict__ X, int cols, float* __restrict__ out, int mode) {
  __shared__ float red[256];
  const int i = blockIdx.x, tid = threadIdx.x;
  float a = 0.f;
  for (int c = tid; c < cols; c += 256) { float v = X[(size_t)i * cols + c]; a += v * v; }
  red[tid] = a; __syncthreads();
  for (int s = 128; s > 0; s >>= 1) { if (tid < s) red[tid] += red[tid + s]; __syncthreads(); }
  if (tid == 0) out[i] = (mode == 0) ? red[0] : sqrtf(red[0]);
}

__global__ void k_dist_rows(float* __restrict__ G, const float* __restrict__ sq, int n,
                            float* __restrict__ partial) {
  __shared__ float red[256];
  const int i = blockIdx.x, tid = threadIdx.x;
  const float si = sq[i];
  float a = 0.f;
  for (int j = tid; j < n; j += 256) {
    float d = fabsf(si + sq[j] - 2.0f * G[(size_t)i * n + j]);
    G[(size_t)i * n + j] = d;
    a += d;
  }
  red[tid] = a; __syncthreads();
  for (int s = 128; s > 0; s >>= 1) { if (tid < s) red[tid] += red[tid + s]; __syncthreads(); }
  if (tid == 0) partial[i] = red[0];
}

__global__ void k_reduce_scaled(const float* __restrict__ p, int n, float* __restrict__ out,
                                float scale) {
  __shared__ float red[256];
  const int tid = threadIdx.x;
  float a = 0.f;
  for (int i = tid; i < n; i += 256) a += p[i];
  red[tid] = a; __syncthreads();
  for (int s = 128; s > 0; s >>= 1) { if (tid < s) red[tid] += red[tid + s]; __syncthreads(); }
  if (tid == 0) out[0] = red[0] * scale;
}

__global__ void k_make_edge(const float* __restrict__ dist, const float* __restrict__ radius,
                            unsigned char* __restrict__ edge, long long count) {
  long long idx = (long long)blockIdx.x * 256 + threadIdx.x;
  if (idx < count) edge[idx] = (dist[idx] < radius[0] * 0.2f) ? 1 : 0;
}

__global__ void k_rowsum_u8_invsqrt(const unsigned char* __restrict__ H, int n,
                                    float* __restrict__ Dv) {
  __shared__ float red[256];
  const int i = blockIdx.x, tid = threadIdx.x;
  float a = 0.f;
  for (int j = tid; j < n; j += 256) a += (float)H[(size_t)i * n + j];
  red[tid] = a; __syncthreads();
  for (int s = 128; s > 0; s >>= 1) { if (tid < s) red[tid] += red[tid + s]; __syncthreads(); }
  if (tid == 0) Dv[i] = rsqrtf(red[0]);
}

__global__ void k_scale_rows_bf16(const float* __restrict__ src, const float* __restrict__ scale,
                                  int rows, int cols, unsigned short* __restrict__ dst) {
  int idx = blockIdx.x * 256 + threadIdx.x;
  if (idx < rows * cols) { int r = idx / cols; dst[idx] = f2bf(src[idx] * scale[r]); }
}

__global__ void k_scale_rows_f32(float* __restrict__ X, const float* __restrict__ scale,
                                 int rows, int cols) {
  int idx = blockIdx.x * 256 + threadIdx.x;
  if (idx < rows * cols) { int r = idx / cols; X[idx] = X[idx] * scale[r]; }
}

__global__ void k_rho_gram(const float* __restrict__ G, const float* __restrict__ nrm,
                           const unsigned char* __restrict__ xx, int n, float sigma,
                           float* __restrict__ rho) {
  __shared__ float red[256];
  const int i = blockIdx.x, tid = threadIdx.x;
  const float ni = nrm[i];
  float a = 0.f;
  for (int j = tid; j < n; j += 256) {
    if (j != i && xx[(size_t)i * n + j]) {
      float c = G[(size_t)i * n + j] / (ni * nrm[j]);
      if (c > sigma) a += c;
    }
  }
  red[tid] = a; __syncthreads();
  for (int s = 128; s > 0; s >>= 1) { if (tid < s) red[tid] += red[tid + s]; __syncthreads(); }
  if (tid == 0) rho[i] = red[0];
}

__global__ void k_mask_dot(const unsigned char* __restrict__ H, const float* __restrict__ v,
                           int n, float* __restrict__ out) {
  __shared__ float red[256];
  const int m = blockIdx.x, tid = threadIdx.x;
  float a = 0.f;
  for (int i = tid; i < n; i += 256)
    if (H[(size_t)m * n + i]) a += v[i];
  red[tid] = a; __syncthreads();
  for (int s = 128; s > 0; s >>= 1) { if (tid < s) red[tid] += red[tid + s]; __syncthreads(); }
  if (tid == 0) out[m] = red[0];
}

__global__ void k_minmax(const float* __restrict__ v, int n, float* __restrict__ omin,
                         float* __restrict__ omax) {
  __shared__ float r1[256], r2[256];
  const int tid = threadIdx.x;
  float mn = 3.0e38f, mx = -3.0e38f;
  for (int i = tid; i < n; i += 256) { mn = fminf(mn, v[i]); mx = fmaxf(mx, v[i]); }
  r1[tid] = mn; r2[tid] = mx; __syncthreads();
  for (int s = 128; s > 0; s >>= 1) {
    if (tid < s) { r1[tid] = fminf(r1[tid], r1[tid + s]); r2[tid] = fmaxf(r2[tid], r2[tid + s]); }
    __syncthreads();
  }
  if (tid == 0) { omin[0] = r1[0]; omax[0] = r2[0]; }
}

__global__ void k_small_proj(const float* __restrict__ W, const float* __restrict__ a,
                             int hid, int d, float* __restrict__ w1, float* __restrict__ w2) {
  int r = blockIdx.x * 256 + threadIdx.x;
  if (r < hid) {
    float s1 = 0.f, s2 = 0.f;
    for (int c = 0; c < d; ++c) { s1 += W[r * d + c] * a[c]; s2 += W[r * d + c] * a[d + c]; }
    w1[r] = s1; w2[r] = s2;
  }
}

__global__ void k_matvec(const float* __restrict__ M, const float* __restrict__ v, int cols,
                         float* __restrict__ out) {
  __shared__ float red[256];
  const int i = blockIdx.x, tid = threadIdx.x;
  float a = 0.f;
  for (int c = tid; c < cols; c += 256) a += M[(size_t)i * cols + c] * v[c];
  red[tid] = a; __syncthreads();
  for (int s = 128; s > 0; s >>= 1) { if (tid < s) red[tid] += red[tid + s]; __syncthreads(); }
  if (tid == 0) out[i] = red[0];
}

__global__ __launch_bounds__(256) void k_rowmat8(const float* __restrict__ Mm,
                                                 const float* __restrict__ W, int cols,
                                                 float* __restrict__ out) {
  __shared__ float red[256 * 8];
  const int i = blockIdx.x, tid = threadIdx.x;
  float acc[8] = {0.f, 0.f, 0.f, 0.f, 0.f, 0.f, 0.f, 0.f};
  for (int k = tid; k < cols; k += 256) {
    const float x = Mm[(size_t)i * cols + k];
    #pragma unroll
    for (int c = 0; c < 8; ++c) acc[c] += x * W[k * 8 + c];
  }
  #pragma unroll
  for (int c = 0; c < 8; ++c) red[tid * 8 + c] = acc[c];
  __syncthreads();
  for (int s = 128; s > 0; s >>= 1) {
    if (tid < s)
      for (int c = 0; c < 8; ++c) red[tid * 8 + c] += red[(tid + s) * 8 + c];
    __syncthreads();
  }
  if (tid < 8) out[(size_t)i * 8 + tid] = red[tid];
}

__global__ void k_smax(const float* __restrict__ sx, const float* __restrict__ se, int n,
                       float* __restrict__ out) {
  __shared__ float r1[256], r2[256];
  const int tid = threadIdx.x;
  float m1 = -3.0e38f, m2 = -3.0e38f;
  for (int i = tid; i < n; i += 256) { m1 = fmaxf(m1, sx[i]); m2 = fmaxf(m2, se[i]); }
  r1[tid] = m1; r2[tid] = m2; __syncthreads();
  for (int s = 128; s > 0; s >>= 1) {
    if (tid < s) { r1[tid] = fmaxf(r1[tid], r1[tid + s]); r2[tid] = fmaxf(r2[tid], r2[tid + s]); }
    __syncthreads();
  }
  if (tid == 0) out[0] = lrelu(r1[0] + r2[0]);
}

__global__ __launch_bounds__(256) void k_att_rows(
    const unsigned char* __restrict__ H8, const float* __restrict__ sx,
    const float* __restrict__ se, const float* __restrict__ rho,
    const float* __restrict__ prmin, const float* __restrict__ prmax,
    const float* __restrict__ psmax, int n, float* __restrict__ baseo,
    float* __restrict__ rowinv) {
  __shared__ float red[256];
  const int i = blockIdx.x, tid = threadIdx.x;
  const float rt = (rho[i] - prmin[0]) / (prmax[0] - prmin[0]) * psmax[0];
  const float sxi = sx[i];
  const unsigned char* Hr = H8 + (size_t)i * n;
  float mx = -3.0e38f;
  for (int j = tid; j < n; j += 256)
    if (Hr[j]) mx = fmaxf(mx, se[j]);
  red[tid] = mx; __syncthreads();
  for (int s = 128; s > 0; s >>= 1) { if (tid < s) red[tid] = fmaxf(red[tid], red[tid + s]); __syncthreads(); }
  const float rowmax = lrelu(sxi + red[0]) + rt;
  __syncthreads();
  float acc = 0.f;
  for (int j = tid; j < n; j += 256)
    if (Hr[j]) acc += expf(lrelu(sxi + se[j]) + rt - rowmax);
  red[tid] = acc; __syncthreads();
  for (int s = 128; s > 0; s >>= 1) { if (tid < s) red[tid] += red[tid + s]; __syncthreads(); }
  if (tid == 0) { baseo[i] = rt - rowmax; rowinv[i] = 1.0f / red[0]; }
}

__global__ __launch_bounds__(256) void k_att_out(
    const unsigned char* __restrict__ H8, const float* __restrict__ sx,
    const float* __restrict__ se, const float* __restrict__ basev,
    const float* __restrict__ rowinv, const float* __restrict__ agg, int n,
    float* __restrict__ out, int ostride, int ocol) {
  __shared__ float red[256 * 8];
  const int k = blockIdx.x, tid = threadIdx.x;
  const float sek = se[k];
  const unsigned char* Hc = H8 + (size_t)k * n;
  float acc[8] = {0.f, 0.f, 0.f, 0.f, 0.f, 0.f, 0.f, 0.f};
  for (int i = tid; i < n; i += 256) {
    if (Hc[i]) {
      float w = expf(lrelu(sx[i] + sek) + basev[i]) * rowinv[i];
      const float* ag = agg + (size_t)i * 8;
      #pragma unroll
      for (int c = 0; c < 8; ++c) acc[c] += w * ag[c];
    }
  }
  #pragma unroll
  for (int c = 0; c < 8; ++c) red[tid * 8 + c] = acc[c];
  __syncthreads();
  for (int s = 128; s > 0; s >>= 1) {
    if (tid < s)
      for (int c = 0; c < 8; ++c) red[tid * 8 + c] += red[(tid + s) * 8 + c];
    __syncthreads();
  }
  if (tid < 8) out[(size_t)k * ostride + ocol + tid] = eluf(red[tid]);
}

__global__ void k_elu_bias(float* __restrict__ X, const float* __restrict__ b, int rows, int cols) {
  int idx = blockIdx.x * 256 + threadIdx.x;
  if (idx < rows * cols) X[idx] = eluf(X[idx] + b[idx % cols]);
}

extern "C" void kernel_launch(void* const* d_in, const int* in_sizes, int n_in,
                              void* d_out, int out_size, void* d_ws, size_t ws_size,
                              hipStream_t stream) {
  const float* X      = (const float*)d_in[0];
  const float* theta  = (const float*)d_in[1];
  const float* Wh_all = (const float*)d_in[2];
  const float* ax_all = (const float*)d_in[3];
  const float* ae_all = (const float*)d_in[4];
  const float* lin_w  = (const float*)d_in[5];
  const float* lin_b  = (const float*)d_in[6];
  const float* W2     = (const float*)d_in[7];
  const float* ax2    = (const float*)d_in[8];
  const float* ae2    = (const float*)d_in[9];
  float* out = (float*)d_out;

  char* basep = (char*)d_ws;
  size_t off = 0;
  auto nxt = [&](size_t bytes) -> void* {
    void* p = basep + off;
    off += (bytes + 255) & ~(size_t)255;
    return p;
  };
  float* BIG            = (float*)nxt((size_t)NN * NN * 4);
  unsigned short* Hbf   = (unsigned short*)nxt((size_t)NN * NN * 2);
  unsigned char* edge8  = (unsigned char*)nxt((size_t)NN * NN);
  unsigned char* H8     = (unsigned char*)nxt((size_t)NN * NN);
  unsigned char* xx8    = (unsigned char*)nxt((size_t)NN * NN);
  unsigned short* Xbf   = (unsigned short*)nxt((size_t)NN * FF * 2);
  unsigned short* DvXbf = (unsigned short*)nxt((size_t)NN * FF * 2);
  float* T1             = (float*)nxt((size_t)NN * FF * 4);
  unsigned short* T1sbf = (unsigned short*)nxt((size_t)NN * FF * 2);
  unsigned short* thbf  = (unsigned short*)nxt((size_t)FF * HIDW * 2);
  float* Ebuf           = (float*)nxt((size_t)NN * HIDW * 4);
  unsigned short* DeEbf = (unsigned short*)nxt((size_t)NN * HIDW * 2);
  float* Xh             = (float*)nxt((size_t)NN * HIDW * 4);
  unsigned short* Xhbf  = (unsigned short*)nxt((size_t)NN * HIDW * 2);
  float* Xc             = (float*)nxt((size_t)NN * 32 * 4);
  unsigned short* Xcbf  = (unsigned short*)nxt((size_t)NN * 32 * 2);
  unsigned short* lwbf  = (unsigned short*)nxt((size_t)32 * HIDW * 2);
  float* Xl             = (float*)nxt((size_t)NN * HIDW * 4);
  unsigned short* Xlbf  = (unsigned short*)nxt((size_t)NN * HIDW * 2);
  float* sq     = (float*)nxt(NN * 4);
  float* Dv     = (float*)nxt(NN * 4);
  float* nrm    = (float*)nxt(NN * 4);
  float* rho_x  = (float*)nxt(NN * 4);
  float* rho_e  = (float*)nxt(NN * 4);
  float* sx     = (float*)nxt(NN * 4);
  float* se     = (float*)nxt(NN * 4);
  float* basev  = (float*)nxt(NN * 4);
  float* rowinv = (float*)nxt(NN * 4);
  float* w1     = (float*)nxt(HIDW * 4);
  float* w2     = (float*)nxt(HIDW * 4);
  float* aggb   = (float*)nxt((size_t)NN * DD * 4);
  float* Enew   = (float*)nxt((size_t)NN * DD * 4);
  float* partial = (float*)nxt(NN * 4);
  float* scal    = (float*)nxt(64 * 4);
  float* smaxp = scal + 5;

  const dim3 B256(256);
  const dim3 gridNN(NN / 128, NN / 64);

  auto attention = [&](const float* Xi, const float* Ek, const float* Wp, const float* av,
                       const float* rho, const float* prmin, const float* prmax,
                       const float* agg, float* outp, int ostride, int ocol) {
    k_small_proj<<<1, B256, 0, stream>>>(Wp, av, HIDW, DD, w1, w2);
    k_matvec<<<NN, B256, 0, stream>>>(Xi, w1, HIDW, sx);
    k_matvec<<<NN, B256, 0, stream>>>(Ek, w2, HIDW, se);
    k_smax<<<1, B256, 0, stream>>>(sx, se, NN, smaxp);
    k_att_rows<<<NN, B256, 0, stream>>>(H8, sx, se, rho, prmin, prmax, smaxp, NN, basev, rowinv);
    k_att_out<<<NN, B256, 0, stream>>>(H8, sx, se, basev, rowinv, agg, NN, outp, ostride, ocol);
  };

  k_f32_to_bf16<<<(NN * FF) / 256, B256, 0, stream>>>(X, Xbf, NN * FF);
  k_f32_to_bf16<<<(FF * HIDW) / 256, B256, 0, stream>>>(theta, thbf, FF * HIDW);
  k_row_sumsq<<<NN, B256, 0, stream>>>(X, FF, sq, 0);
  gemm_bf16_kernel<true><<<gridNN, B256, 0, stream>>>(Xbf, Xbf, BIG, NN, NN, FF);
  k_dist_rows<<<NN, B256, 0, stream>>>(BIG, sq, NN, partial);
  k_reduce_scaled<<<1, B256, 0, stream>>>(partial, NN, scal, 1.0f / ((float)NN * (float)NN));
  k_make_edge<<<(NN * NN) / 256, B256, 0, stream>>>(BIG, scal, edge8, (long long)NN * NN);
  gemm_iu8_bool_kernel<<<gridNN, B256, 0, stream>>>(edge8, edge8, H8, Hbf, NN, 0);
  k_rowsum_u8_invsqrt<<<NN, B256, 0, stream>>>(H8, NN, Dv);
  k_scale_rows_bf16<<<(NN * FF) / 256, B256, 0, stream>>>(X, Dv, NN, FF, DvXbf);
  gemm_bf16_kernel<false><<<dim3(FF / 128, NN / 64), B256, 0, stream>>>(Hbf, DvXbf, T1, NN, FF, NN);
  k_scale_rows_bf16<<<(NN * FF) / 256, B256, 0, stream>>>(T1, Dv, NN, FF, T1sbf);
  gemm_bf16_kernel<false><<<dim3(HIDW / 128, NN / 64), B256, 0, stream>>>(T1sbf, thbf, Ebuf, NN, HIDW, FF);
  k_scale_rows_bf16<<<(NN * HIDW) / 256, B256, 0, stream>>>(Ebuf, Dv, NN, HIDW, DeEbf);
  gemm_bf16_kernel<false><<<dim3(HIDW / 128, NN / 64), B256, 0, stream>>>(Hbf, DeEbf, Xh, NN, HIDW, NN);
  k_scale_rows_f32<<<(NN * HIDW) / 256, B256, 0, stream>>>(Xh, Dv, NN, HIDW);

  k_f32_to_bf16<<<(NN * HIDW) / 256, B256, 0, stream>>>(Xh, Xhbf, NN * HIDW);
  k_row_sumsq<<<NN, B256, 0, stream>>>(Xh, HIDW, nrm, 1);
  gemm_iu8_bool_kernel<<<gridNN, B256, 0, stream>>>(H8, H8, xx8, (unsigned short*)nullptr, NN, 1);
  gemm_bf16_kernel<true><<<gridNN, B256, 0, stream>>>(Xhbf, Xhbf, BIG, NN, NN, HIDW);
  k_rho_gram<<<NN, B256, 0, stream>>>(BIG, nrm, xx8, NN, SIG, rho_x);
  k_mask_dot<<<NN, B256, 0, stream>>>(H8, rho_x, NN, rho_e);
  k_minmax<<<1, B256, 0, stream>>>(rho_x, NN, scal + 1, scal + 2);
  k_minmax<<<1, B256, 0, stream>>>(rho_e, NN, scal + 3, scal + 4);

  for (int h = 0; h < NHEADS; ++h) {
    const float* Wp  = Wh_all + (size_t)h * HIDW * DD;
    const float* axp = ax_all + (size_t)h * 2 * DD;
    const float* aep = ae_all + (size_t)h * 2 * DD;
    k_rowmat8<<<NN, B256, 0, stream>>>(Xh, Wp, HIDW, aggb);
    attention(Xh, Ebuf, Wp, axp, rho_x, scal + 1, scal + 2, aggb, Enew, DD, 0);
    attention(Ebuf, Xh, Wp, aep, rho_e, scal + 3, scal + 4, Enew, Xc, 32, h * DD);
  }

  k_f32_to_bf16<<<(NN * 32) / 256, B256, 0, stream>>>(Xc, Xcbf, NN * 32);
  k_f32_to_bf16<<<(32 * HIDW) / 256, B256, 0, stream>>>(lin_w, lwbf, 32 * HIDW);
  gemm_bf16_kernel<false><<<dim3(HIDW / 128, NN / 64), B256, 0, stream>>>(Xcbf, lwbf, Xl, NN, HIDW, 32);
  k_elu_bias<<<(NN * HIDW) / 256, B256, 0, stream>>>(Xl, lin_b, NN, HIDW);

  k_f32_to_bf16<<<(NN * HIDW) / 256, B256, 0, stream>>>(Xl, Xlbf, NN * HIDW);
  k_row_sumsq<<<NN, B256, 0, stream>>>(Xl, HIDW, nrm, 1);
  gemm_bf16_kernel<true><<<gridNN, B256, 0, stream>>>(Xlbf, Xlbf, BIG, NN, NN, HIDW);
  k_rho_gram<<<NN, B256, 0, stream>>>(BIG, nrm, xx8, NN, SIG, rho_x);
  k_mask_dot<<<NN, B256, 0, stream>>>(H8, rho_x, NN, rho_e);
  k_minmax<<<1, B256, 0, stream>>>(rho_x, NN, scal + 1, scal + 2);
  k_minmax<<<1, B256, 0, stream>>>(rho_e, NN, scal + 3, scal + 4);
  k_rowmat8<<<NN, B256, 0, stream>>>(Xl, W2, HIDW, aggb);
  attention(Xl, Ebuf, W2, ax2, rho_x, scal + 1, scal + 2, aggb, Enew, DD, 0);
  attention(Ebuf, Xl, W2, ae2, rho_e, scal + 3, scal + 4, Enew, out, DD, 0);
}
